// LeapfrogIntegrator_70635032150167
// MI455X (gfx1250) — compile-verified, matrix operand order checked by reading, not run on hardware
//
#include <hip/hip_runtime.h>
#include <hip/hip_bf16.h>

// ---------------------------------------------------------------------------
// Fused leapfrog integrator with low-rank Christoffel term, MI455X (gfx1250).
//   gamma(v,x) = (tanh(x@U) @ W) * v * v
// bf16 WMMA (16x16x32, f32 accum) for all four GEMMs, fully fused per 32-row
// batch tile: x/v/force read once from HBM, x_new/v_new written once ->
// bandwidth-bound at the 23.3 TB/s roofline (~29us floor; bf16 matrix math
// is far below that, f32 matrix math would not be).
//
// Round-2 changes:
//  * tanh via gfx1250 hardware V_TANH_F32 (guarded __has_builtin), removing
//    the divergent ~60-op ocml expansion seen in round-1 disassembly.
//  * U and W pre-packed once into d_ws as bf16 in exact WMMA B-fragment lane
//    order -> each B fragment is one coalesced 32B vector load instead of 16
//    strided dword gathers (and halves B-side L2 traffic).
// ---------------------------------------------------------------------------

typedef __attribute__((ext_vector_type(16))) __bf16 v16bf;
typedef __attribute__((ext_vector_type(8)))  __bf16 v8bf;
typedef __attribute__((ext_vector_type(8)))  float  v8f;

#define B_ROWS 32768
#define D_DIM  1024
#define R_DIM  128
#define BM     32      // batch rows per workgroup (2 WMMA m-tiles)
#define XS     1032    // padded bf16 LDS stride for x_new tile (16B-aligned rows)
#define TS     136     // padded bf16 LDS stride for tanh tile  (16B-aligned rows)
#define EFF_DT 0.1f    // DT * DT_SCALE

#define UPK_ELEMS (D_DIM * R_DIM)           // 131072 bf16
#define WPK_ELEMS (R_DIM * D_DIM)           // 131072 bf16
#define PACK_BYTES ((UPK_ELEMS + WPK_ELEMS) * 2)  // 512 KB

// ---- hardware tanh (TRANS32 V_TANH_F32 on gfx1250), branch-free fallback ----
static __device__ __forceinline__ float fast_tanhf(float x) {
#if __has_builtin(__builtin_amdgcn_tanhf)
  return __builtin_amdgcn_tanhf(x);
#elif __has_builtin(__builtin_amdgcn_tanh_f32)
  return __builtin_amdgcn_tanh_f32(x);
#else
  // branch-free identity: tanh(x) = 1 - 2/(e^{2x}+1); clamp keeps exp finite.
  float xc = fminf(fmaxf(x, -15.0f), 15.0f);
  float e  = __expf(2.0f * xc);
  return (e - 1.0f) / (e + 1.0f);
#endif
}

static __device__ __forceinline__ v8f wmma_bf16(v16bf a, v16bf b, v8f c) {
  // D = A(16x32) * B(32x16) + C, f32 accumulate
  return __builtin_amdgcn_wmma_f32_16x16x32_bf16(
      /*neg_a=*/false, a, /*neg_b=*/false, b,
      /*c_mod=*/(short)0, c, /*reuse_a=*/false, /*reuse_b=*/false);
}

// A-fragment (16x32, bf16) from a row-major f32 matrix.
// Layout per ISA: lanes 0-15 hold row M=l16, K={0..7,16..23}; lanes 16-31 hold
// K={8..15,24..31}. Two 8-float chunks per lane -> 4x float4 loads + cvt.
static __device__ __forceinline__ v16bf
a_frag_global(const float* __restrict__ row0, int stride, int kb, int l16, int half) {
  const float* r = row0 + (size_t)l16 * stride + kb + (half ? 8 : 0);
  float4 f0 = *(const float4*)(r + 0);
  float4 f1 = *(const float4*)(r + 4);
  float4 f2 = *(const float4*)(r + 16);
  float4 f3 = *(const float4*)(r + 20);
  v16bf a;
  a[0]  = (__bf16)f0.x; a[1]  = (__bf16)f0.y; a[2]  = (__bf16)f0.z; a[3]  = (__bf16)f0.w;
  a[4]  = (__bf16)f1.x; a[5]  = (__bf16)f1.y; a[6]  = (__bf16)f1.z; a[7]  = (__bf16)f1.w;
  a[8]  = (__bf16)f2.x; a[9]  = (__bf16)f2.y; a[10] = (__bf16)f2.z; a[11] = (__bf16)f2.w;
  a[12] = (__bf16)f3.x; a[13] = (__bf16)f3.y; a[14] = (__bf16)f3.z; a[15] = (__bf16)f3.w;
  return a;
}

// A-fragment from a row-major bf16 LDS tile (rows 16B-aligned) -> 2x ds_load_b128.
static __device__ __forceinline__ v16bf
a_frag_lds(const __bf16* buf, int stride, int mrow, int kb, int l16, int half) {
  const __bf16* r = buf + (mrow + l16) * stride + kb + (half ? 8 : 0);
  v8bf lo = *(const v8bf*)(r);
  v8bf hi = *(const v8bf*)(r + 16);
  v16bf a;
#pragma unroll
  for (int e = 0; e < 8; ++e) { a[e] = lo[e]; a[8 + e] = hi[e]; }
  return a;
}

// B-fragment (32x16, bf16) gathered from a row-major f32 matrix (fallback path).
static __device__ __forceinline__ v16bf
b_frag_global(const float* __restrict__ base, int stride, int kb, int nb, int l16, int half) {
  const int c0 = kb + (half ? 8 : 0);
  const float* p = base + (size_t)c0 * stride + nb + l16;
  v16bf b;
#pragma unroll
  for (int e = 0; e < 8; ++e) b[e] = (__bf16)p[e * stride];
  p += 16 * stride;
#pragma unroll
  for (int e = 0; e < 8; ++e) b[8 + e] = (__bf16)p[e * stride];
  return b;
}

// B-fragment from pre-packed bf16 tiles: layout [ntile][ktile][lane][16].
// One 32B contiguous load per lane, fully coalesced across the wave.
static __device__ __forceinline__ v16bf
b_frag_packed(const __bf16* __restrict__ pk, int ktc, int kt, int nt, int lid) {
  const __bf16* p = pk + ((((size_t)nt * ktc + kt) * 32 + lid) << 4);
  return *(const v16bf*)p;
}

// ---- pack kernel: U (DxR) and W (RxD) f32 -> bf16 WMMA-B tile order in ws ----
extern "C" __global__ void __launch_bounds__(256)
pack_b_kernel(const float* __restrict__ U, const float* __restrict__ W,
              __bf16* __restrict__ ws) {
  const int p = blockIdx.x * blockDim.x + threadIdx.x;  // 0 .. 262143
  const int e    = p & 15;
  const int lid  = (p >> 4) & 31;
  const int tile = p >> 9;
  const int half = lid >> 4;
  const int l16  = lid & 15;
  const int kofs = half * 8 + ((e < 8) ? e : (e + 8));  // K within 32-chunk
  if (p < UPK_ELEMS) {                // U: K = D_DIM, N = R_DIM, ktiles = 32
    const int kt = tile & 31, nt = tile >> 5;
    const int k = kt * 32 + kofs, n = nt * 16 + l16;
    ws[p] = (__bf16)U[k * R_DIM + n];
  } else {                            // W: K = R_DIM, N = D_DIM, ktiles = 4
    const int q = p - UPK_ELEMS;
    const int t2 = q >> 9;
    const int kt = t2 & 3, nt = t2 >> 2;
    const int k = kt * 32 + kofs, n = nt * 16 + l16;
    ws[p] = (__bf16)W[k * D_DIM + n];
  }
}

template <bool PACKED>
__global__ void __launch_bounds__(256)
leapfrog_wmma_kernel(const float* __restrict__ x, const float* __restrict__ v,
                     const float* __restrict__ frc, const float* __restrict__ U,
                     const float* __restrict__ W, const __bf16* __restrict__ Upk,
                     const __bf16* __restrict__ Wpk, float* __restrict__ xout,
                     float* __restrict__ vout) {
  extern __shared__ char smem[];
  __bf16* Xb = reinterpret_cast<__bf16*>(smem);                          // [BM][XS] bf16 x_new
  __bf16* Tb = reinterpret_cast<__bf16*>(smem + (size_t)BM * XS * 2);    // [BM][TS] bf16 tanh(.)

  const int tid  = threadIdx.x;
  const int wave = tid >> 5;        // 8 waves
  const int lid  = tid & 31;
  const int l16  = lid & 15;
  const int half = lid >> 4;
  const int rowbase = blockIdx.x * BM;

  // ---------------- Phase 1: T1 = tanh(x_tile @ U) ----------------
  {
    const int n0 = wave * 16;
    v8f acc0 = {}; v8f acc1 = {};
    for (int kb = 0; kb < D_DIM; kb += 32) {
      __builtin_prefetch(x + (size_t)(rowbase + l16) * D_DIM + kb + 64, 0, 0);
      v16bf bf = PACKED ? b_frag_packed(Upk, 32, kb >> 5, wave, lid)
                        : b_frag_global(U, R_DIM, kb, n0, l16, half);
      v16bf a0 = a_frag_global(x + (size_t)(rowbase + 0)  * D_DIM, D_DIM, kb, l16, half);
      v16bf a1 = a_frag_global(x + (size_t)(rowbase + 16) * D_DIM, D_DIM, kb, l16, half);
      acc0 = wmma_bf16(a0, bf, acc0);
      acc1 = wmma_bf16(a1, bf, acc1);
    }
    // C/D layout: VGPR j -> row M = j + 8*half (+16 per m-tile), col = lane%16.
#pragma unroll
    for (int j = 0; j < 8; ++j) {
      Tb[(0  + j + half * 8) * TS + n0 + l16] = (__bf16)fast_tanhf(acc0[j]);
      Tb[(16 + j + half * 8) * TS + n0 + l16] = (__bf16)fast_tanhf(acc1[j]);
    }
  }
  __syncthreads();

  // ---------- Phase 2: G1 = T1 @ W, fused half-kick + drift ----------
  {
    const int ncol0 = wave * 128;
    for (int s = 0; s < 8; ++s) {
      const int n0 = ncol0 + s * 16;
      v8f acc0 = {}; v8f acc1 = {};
#pragma unroll
      for (int kb = 0; kb < R_DIM; kb += 32) {
        v16bf bf = PACKED ? b_frag_packed(Wpk, 4, kb >> 5, n0 >> 4, lid)
                          : b_frag_global(W, D_DIM, kb, n0, l16, half);
        v16bf a0 = a_frag_lds(Tb, TS, 0,  kb, l16, half);
        v16bf a1 = a_frag_lds(Tb, TS, 16, kb, l16, half);
        acc0 = wmma_bf16(a0, bf, acc0);
        acc1 = wmma_bf16(a1, bf, acc1);
      }
#pragma unroll
      for (int j = 0; j < 8; ++j) {
#pragma unroll
        for (int mt = 0; mt < 2; ++mt) {
          const int   m   = mt * 16 + j + half * 8;
          const int   idx = (rowbase + m) * D_DIM + n0 + l16;
          const float g   = (mt == 0) ? acc0[j] : acc1[j];
          const float vv  = v[idx];
          const float ff  = frc[idx];
          const float xx  = x[idx];
          const float vh  = vv + 0.5f * EFF_DT * (ff - g * vv * vv);   // half-kick
          const float xn  = xx + EFF_DT * vh;                          // drift
          xout[idx] = xn;
          vout[idx] = vh;                         // park v_half (same lane re-reads it)
          Xb[m * XS + n0 + l16] = (__bf16)xn;     // bf16 copy for GEMM3
        }
      }
    }
  }
  __syncthreads();   // Tb reads done; Xb fully written

  // ---------------- Phase 3: T2 = tanh(x_new @ U) ----------------
  {
    const int n0 = wave * 16;
    v8f acc0 = {}; v8f acc1 = {};
    for (int kb = 0; kb < D_DIM; kb += 32) {
      v16bf bf = PACKED ? b_frag_packed(Upk, 32, kb >> 5, wave, lid)
                        : b_frag_global(U, R_DIM, kb, n0, l16, half);
      v16bf a0 = a_frag_lds(Xb, XS, 0,  kb, l16, half);
      v16bf a1 = a_frag_lds(Xb, XS, 16, kb, l16, half);
      acc0 = wmma_bf16(a0, bf, acc0);
      acc1 = wmma_bf16(a1, bf, acc1);
    }
#pragma unroll
    for (int j = 0; j < 8; ++j) {
      Tb[(0  + j + half * 8) * TS + n0 + l16] = (__bf16)fast_tanhf(acc0[j]);
      Tb[(16 + j + half * 8) * TS + n0 + l16] = (__bf16)fast_tanhf(acc1[j]);
    }
  }
  __syncthreads();

  // ---------- Phase 4: G2 = T2 @ W, final half-kick ----------
  {
    const int ncol0 = wave * 128;
    for (int s = 0; s < 8; ++s) {
      const int n0 = ncol0 + s * 16;
      v8f acc0 = {}; v8f acc1 = {};
#pragma unroll
      for (int kb = 0; kb < R_DIM; kb += 32) {
        v16bf bf = PACKED ? b_frag_packed(Wpk, 4, kb >> 5, n0 >> 4, lid)
                          : b_frag_global(W, D_DIM, kb, n0, l16, half);
        v16bf a0 = a_frag_lds(Tb, TS, 0,  kb, l16, half);
        v16bf a1 = a_frag_lds(Tb, TS, 16, kb, l16, half);
        acc0 = wmma_bf16(a0, bf, acc0);
        acc1 = wmma_bf16(a1, bf, acc1);
      }
#pragma unroll
      for (int j = 0; j < 8; ++j) {
#pragma unroll
        for (int mt = 0; mt < 2; ++mt) {
          const int   m   = mt * 16 + j + half * 8;
          const int   idx = (rowbase + m) * D_DIM + n0 + l16;
          const float g   = (mt == 0) ? acc0[j] : acc1[j];
          const float vh  = vout[idx];            // same-lane store->load, in order
          const float ff  = frc[idx];
          vout[idx] = vh + 0.5f * EFF_DT * (ff - g * vh * vh);   // final half-kick
        }
      }
    }
  }
}

extern "C" void kernel_launch(void* const* d_in, const int* in_sizes, int n_in,
                              void* d_out, int out_size, void* d_ws, size_t ws_size,
                              hipStream_t stream) {
  (void)in_sizes; (void)n_in; (void)out_size;
  const float* x   = (const float*)d_in[0];
  const float* v   = (const float*)d_in[1];
  const float* frc = (const float*)d_in[2];
  const float* U   = (const float*)d_in[3];
  const float* W   = (const float*)d_in[4];
  float* xout = (float*)d_out;                               // x_new
  float* vout = xout + (size_t)B_ROWS * D_DIM;               // v_new (also v_half scratch)

  const size_t shmem = (size_t)BM * XS * 2 + (size_t)BM * TS * 2;  // ~73 KB
  dim3 grid(B_ROWS / BM);   // 1024 workgroups
  dim3 block(256);          // 8 wave32s

  const bool packed = (d_ws != nullptr) && (ws_size >= (size_t)PACK_BYTES);
  if (packed) {
    __bf16* ws = (__bf16*)d_ws;
    pack_b_kernel<<<dim3((UPK_ELEMS + WPK_ELEMS) / 256), dim3(256), 0, stream>>>(U, W, ws);
    leapfrog_wmma_kernel<true><<<grid, block, shmem, stream>>>(
        x, v, frc, U, W, ws, ws + UPK_ELEMS, xout, vout);
  } else {
    leapfrog_wmma_kernel<false><<<grid, block, shmem, stream>>>(
        x, v, frc, U, W, nullptr, nullptr, xout, vout);
  }
}
